// RelPosMultiHeadSelfAttention_1812476199766
// MI455X (gfx1250) — matrix operand order repaired and checked by reading, not compiled, not run on hardware
//
#include <hip/hip_runtime.h>

// Problem constants (Transformer-XL rel-pos attention)
#define SS  2048
#define BB  2
#define MMEM 512
#define HH  12
#define DD  64
#define DMM 768
#define KK  2560   // SS + MMEM

typedef __bf16 bf16_t;
typedef __attribute__((ext_vector_type(16))) __bf16 v16bf;
typedef __attribute__((ext_vector_type(8)))  __bf16 v8bf;
typedef __attribute__((ext_vector_type(8)))  float  v8f;

// ---------------------------------------------------------------- helpers

__device__ __forceinline__ v8f zero_v8f() {
  v8f c;
#pragma unroll
  for (int i = 0; i < 8; ++i) c[i] = 0.f;
  return c;
}

__device__ __forceinline__ v8f wmma_bf16(v16bf a, v16bf b, v8f c) {
  // D = A(16x32 bf16) x B(32x16 bf16) + C(16x16 f32)
  return __builtin_amdgcn_wmma_f32_16x16x32_bf16(
      false, a, false, b, (short)0, c, false, false);
}

// A-fragment, 16x32 bf16, from row-major source (row stride lda elements).
// ISA layout: lane<16  -> row=lane,    K = {kb+0..7,  kb+16..23}
//             lane>=16 -> row=lane-16, K = {kb+8..15, kb+24..31}
__device__ __forceinline__ v16bf load_afrag(const bf16_t* base, int lda, int kb) {
  int lane = threadIdx.x & 31;
  int hl = lane >> 4;
  int m  = lane & 15;
  const bf16_t* p = base + (size_t)m * lda + kb + hl * 8;
  v8bf r0 = *(const v8bf*)(p);
  v8bf r1 = *(const v8bf*)(p + 16);
  v16bf a;
#pragma unroll
  for (int i = 0; i < 8; ++i) { a[i] = r0[i]; a[i + 8] = r1[i]; }
  return a;
}

// A-fragment from f32 LDS staging buffer (16x32 row-major, lda = 32)
__device__ __forceinline__ v16bf load_afrag_f32(const float* base) {
  int lane = threadIdx.x & 31;
  int hl = lane >> 4;
  int m  = lane & 15;
  const float* p = base + m * 32 + hl * 8;
  v16bf a;
#pragma unroll
  for (int i = 0; i < 8; ++i) {
    a[i]     = (bf16_t)p[i];
    a[i + 8] = (bf16_t)p[16 + i];
  }
  return a;
}

// B-fragment, 32x16 bf16, B[k][n] = src[n*ld + k]; base -> element [n=0][k=0].
// ISA layout: lane<16 -> col n=lane, K={0..7,16..23}; lane>=16 -> col n=lane-16, K={8..15,24..31}
__device__ __forceinline__ v16bf load_bfrag(const bf16_t* base, int ld) {
  int lane = threadIdx.x & 31;
  int hl = lane >> 4;
  int n  = lane & 15;
  const bf16_t* p = base + (size_t)n * ld + hl * 8;
  v8bf r0 = *(const v8bf*)(p);
  v8bf r1 = *(const v8bf*)(p + 16);
  v16bf b;
#pragma unroll
  for (int i = 0; i < 8; ++i) { b[i] = r0[i]; b[i + 8] = r1[i]; }
  return b;
}

// one wave computes a 16x64 f32 tile of A(16xk) * W(colsxk)^T
__device__ __forceinline__ void gemm16x64(const bf16_t* A, int lda,
                                          const bf16_t* Wc, int ldw,
                                          int kdim, v8f c[4]) {
  for (int kb = 0; kb < kdim; kb += 32) {
    v16bf a = load_afrag(A, lda, kb);
    __builtin_prefetch(A + kb + 64, 0, 1);
#pragma unroll
    for (int t = 0; t < 4; ++t) {
      v16bf b = load_bfrag(Wc + (size_t)(t * 16) * ldw + kb, ldw);
      c[t] = wmma_bf16(a, b, c[t]);
    }
  }
}

// ---------------------------------------------------------------- kernels

__global__ void cvt_f32_bf16(const float* __restrict__ src,
                             bf16_t* __restrict__ dst, int n) {
  int i = blockIdx.x * 256 + threadIdx.x;
  if (i < n) dst[i] = (bf16_t)src[i];
}

// qx = xin @ Wq^T + bq ; write q_u = qx+u and q_v = qx+v, bf16 row-major [4096][768]
__global__ void qproj_kernel(const bf16_t* __restrict__ xin,
                             const bf16_t* __restrict__ wq,
                             const float* __restrict__ bq,
                             const float* __restrict__ uvec,
                             const float* __restrict__ vvec,
                             bf16_t* __restrict__ q_u,
                             bf16_t* __restrict__ q_v) {
  int r0 = blockIdx.x * 16, c0 = blockIdx.y * 64;
  v8f c[4] = {zero_v8f(), zero_v8f(), zero_v8f(), zero_v8f()};
  gemm16x64(xin + (size_t)r0 * DMM, DMM, wq + (size_t)c0 * DMM, DMM, DMM, c);
  int lane = threadIdx.x & 31, hl = lane >> 4, l15 = lane & 15;
#pragma unroll
  for (int t = 0; t < 4; ++t) {
#pragma unroll
    for (int e = 0; e < 8; ++e) {
      int r = r0 + e + hl * 8;
      int col = c0 + t * 16 + l15;           // col = h*64 + d
      float acc = c[t][e] + bq[col];
      q_u[(size_t)r * DMM + col] = (bf16_t)(acc + uvec[col]);
      q_v[(size_t)r * DMM + col] = (bf16_t)(acc + vvec[col]);
    }
  }
}

// kx = kv @ Wke^T + b ; scatter to kxb[b][h][j][d]   (rows r = j*B + b)
__global__ void keproj_kernel(const bf16_t* __restrict__ kv,
                              const bf16_t* __restrict__ wke,
                              const float* __restrict__ bke,
                              bf16_t* __restrict__ kxb) {
  int r0 = blockIdx.x * 16, c0 = blockIdx.y * 64;
  v8f c[4] = {zero_v8f(), zero_v8f(), zero_v8f(), zero_v8f()};
  gemm16x64(kv + (size_t)r0 * DMM, DMM, wke + (size_t)c0 * DMM, DMM, DMM, c);
  int lane = threadIdx.x & 31, hl = lane >> 4, l15 = lane & 15;
#pragma unroll
  for (int t = 0; t < 4; ++t) {
#pragma unroll
    for (int e = 0; e < 8; ++e) {
      int r = r0 + e + hl * 8;
      int col = c0 + t * 16 + l15;
      int j = r >> 1, b = r & 1, h = col >> 6, d = col & 63;
      kxb[(((size_t)b * HH + h) * KK + j) * DD + d] = (bf16_t)(c[t][e] + bke[col]);
    }
  }
}

// vx = kv @ Wv^T + b ; scatter TRANSPOSED to vxT[b][h][d][j]
__global__ void vproj_kernel(const bf16_t* __restrict__ kv,
                             const bf16_t* __restrict__ wv,
                             const float* __restrict__ bv,
                             bf16_t* __restrict__ vxT) {
  int r0 = blockIdx.x * 16, c0 = blockIdx.y * 64;
  v8f c[4] = {zero_v8f(), zero_v8f(), zero_v8f(), zero_v8f()};
  gemm16x64(kv + (size_t)r0 * DMM, DMM, wv + (size_t)c0 * DMM, DMM, DMM, c);
  int lane = threadIdx.x & 31, hl = lane >> 4, l15 = lane & 15;
#pragma unroll
  for (int t = 0; t < 4; ++t) {
#pragma unroll
    for (int e = 0; e < 8; ++e) {
      int r = r0 + e + hl * 8;
      int col = c0 + t * 16 + l15;
      int j = r >> 1, b = r & 1, h = col >> 6, d = col & 63;
      vxT[(((size_t)b * HH + h) * DD + d) * KK + j] = (bf16_t)(c[t][e] + bv[col]);
    }
  }
}

// kr = rel @ Wkr^T + b ; write krb[h][jj][d]   (rows r = jj)
__global__ void krproj_kernel(const bf16_t* __restrict__ relb,
                              const bf16_t* __restrict__ wkr,
                              const float* __restrict__ bkr,
                              bf16_t* __restrict__ krb) {
  int r0 = blockIdx.x * 16, c0 = blockIdx.y * 64;
  v8f c[4] = {zero_v8f(), zero_v8f(), zero_v8f(), zero_v8f()};
  gemm16x64(relb + (size_t)r0 * DMM, DMM, wkr + (size_t)c0 * DMM, DMM, DMM, c);
  int lane = threadIdx.x & 31, hl = lane >> 4, l15 = lane & 15;
#pragma unroll
  for (int t = 0; t < 4; ++t) {
#pragma unroll
    for (int e = 0; e < 8; ++e) {
      int r = r0 + e + hl * 8;
      int col = c0 + t * 16 + l15;
      int h = col >> 6, d = col & 63;
      krb[((size_t)h * KK + r) * DD + d] = (bf16_t)(c[t][e] + bkr[col]);
    }
  }
}

// out = attnA @ Wo^T + bo  (f32 output [B,S,DM] row-major)
__global__ void oproj_kernel(const bf16_t* __restrict__ attnA,
                             const bf16_t* __restrict__ wo,
                             const float* __restrict__ bo,
                             float* __restrict__ out) {
  int r0 = blockIdx.x * 16, c0 = blockIdx.y * 64;
  v8f c[4] = {zero_v8f(), zero_v8f(), zero_v8f(), zero_v8f()};
  gemm16x64(attnA + (size_t)r0 * DMM, DMM, wo + (size_t)c0 * DMM, DMM, DMM, c);
  int lane = threadIdx.x & 31, hl = lane >> 4, l15 = lane & 15;
#pragma unroll
  for (int t = 0; t < 4; ++t) {
#pragma unroll
    for (int e = 0; e < 8; ++e) {
      int r = r0 + e + hl * 8;
      int col = c0 + t * 16 + l15;
      out[(size_t)r * DMM + col] = c[t][e] + bo[col];
    }
  }
}

// --------------------------------------------------- fused flash attention
// grid = (S/16, H, B), block = 256 (8 waves). Dynamic LDS:
//   bdsh[16][K] f32 (shifted bd table, 160KB) | pst[8][16][32] | ost[8][16][64]
//   | mst[128] | lst[128]
__global__ void flash_kernel(const bf16_t* __restrict__ q_u,
                             const bf16_t* __restrict__ q_v,
                             const bf16_t* __restrict__ kxb,
                             const bf16_t* __restrict__ vxT,
                             const bf16_t* __restrict__ krb,
                             const int* __restrict__ amask,
                             bf16_t* __restrict__ attnA) {
  extern __shared__ float smem[];
  float* bdsh = smem;                        // 16*KK
  float* pst  = bdsh + 16 * KK;              // 8*16*32
  float* ost  = pst + 8 * 16 * 32;           // 8*16*64
  float* mst  = ost + 8 * 16 * 64;           // 128
  float* lst  = mst + 128;                   // 128

  const int i0 = blockIdx.x * 16;
  const int h  = blockIdx.y;
  const int b  = blockIdx.z;
  const int tid = threadIdx.x;
  const int w   = tid >> 5;
  const int lane = tid & 31;
  const int hl = lane >> 4, l15 = lane & 15;

  const bf16_t* krh = krb + (size_t)h * KK * DD;

  // ---- phase 1: bd = (q+v) @ kr^T, rel-shifted straight into LDS ----
  {
    const bf16_t* qvb = q_v + ((size_t)i0 * BB + b) * DMM + h * DD;
    v16bf av0 = load_afrag(qvb, BB * DMM, 0);
    v16bf av1 = load_afrag(qvb, BB * DMM, 32);
    for (int t = w; t < KK / 16; t += 8) {
      int jj0 = t * 16;
      v8f cc = zero_v8f();
      cc = wmma_bf16(av0, load_bfrag(krh + (size_t)jj0 * DD + 0, DD), cc);
      cc = wmma_bf16(av1, load_bfrag(krh + (size_t)jj0 * DD + 32, DD), cc);
      int jj = jj0 + l15;
#pragma unroll
      for (int e = 0; e < 8; ++e) {
        int m = e + hl * 8;
        int i = i0 + m;
        int thr = SS - i - 1;
        int it, jt;
        if (jj >= thr) { it = m;     jt = jj - thr; }
        else           { it = m - 1; jt = jj + MMEM + i + 1; }
        if (it >= 0) bdsh[(size_t)it * KK + jt] = cc[e];
      }
    }
    // zero column (the relative-shift's inserted zero)
    if (tid < 16) {
      int i = i0 + tid, jz = MMEM + i + 1;
      if (jz < KK) bdsh[(size_t)tid * KK + jz] = 0.f;
    }
    // wrap-source row i0+16 for output row i0+15 (VALU)
    int ix = i0 + 16;
    if (ix < SS) {
      const bf16_t* qrow = q_v + ((size_t)ix * BB + b) * DMM + h * DD;
      int count = SS - i0 - 17;
      for (int jj2 = tid; jj2 < count; jj2 += 256) {
        const bf16_t* kr2 = krh + (size_t)jj2 * DD;
        float acc = 0.f;
#pragma unroll
        for (int d = 0; d < DD; ++d) acc += (float)qrow[d] * (float)kr2[d];
        bdsh[(size_t)15 * KK + (MMEM + i0 + 17 + jj2)] = acc;
      }
    }
  }
  __syncthreads();

  // ---- phase 2: online-softmax flash loop (each wave owns K/8 cols) ----
  const bf16_t* qub = q_u + ((size_t)i0 * BB + b) * DMM + h * DD;
  v16bf au0 = load_afrag(qub, BB * DMM, 0);
  v16bf au1 = load_afrag(qub, BB * DMM, 32);
  const bf16_t* kxh = kxb + ((size_t)b * HH + h) * (size_t)KK * DD;
  const bf16_t* vth = vxT + ((size_t)b * HH + h) * (size_t)DD * KK;
  const int* maskb = amask + (size_t)b * SS * SS;

  float m8[8], l8[8];
#pragma unroll
  for (int e = 0; e < 8; ++e) { m8[e] = -1e30f; l8[e] = 0.f; }
  v8f o0 = zero_v8f(), o1 = zero_v8f(), o2 = zero_v8f(), o3 = zero_v8f();
  float* pw = pst + w * 16 * 32;

  const int jbeg = w * (KK / 8);
  for (int j0 = jbeg; j0 < jbeg + (KK / 8); j0 += 32) {
    v8f s0 = zero_v8f(), s1 = zero_v8f();
    s0 = wmma_bf16(au0, load_bfrag(kxh + (size_t)j0 * DD + 0, DD), s0);
    s0 = wmma_bf16(au1, load_bfrag(kxh + (size_t)j0 * DD + 32, DD), s0);
    s1 = wmma_bf16(au0, load_bfrag(kxh + (size_t)(j0 + 16) * DD + 0, DD), s1);
    s1 = wmma_bf16(au1, load_bfrag(kxh + (size_t)(j0 + 16) * DD + 32, DD), s1);

#pragma unroll
    for (int e = 0; e < 8; ++e) {
      int m = e + hl * 8;
      int i = i0 + m;
      int jA = j0 + l15, jB = j0 + 16 + l15;
      float vA = (s0[e] + bdsh[(size_t)m * KK + jA]) * 0.125f;
      float vB = (s1[e] + bdsh[(size_t)m * KK + jB]) * 0.125f;
      if (jA >= MMEM && maskb[(size_t)i * SS + (jA - MMEM)] == 0) vA -= 1e9f;
      if (jB >= MMEM && maskb[(size_t)i * SS + (jB - MMEM)] == 0) vB -= 1e9f;
      float t = fmaxf(vA, vB);
      t = fmaxf(t, __shfl_xor(t, 1));
      t = fmaxf(t, __shfl_xor(t, 2));
      t = fmaxf(t, __shfl_xor(t, 4));
      t = fmaxf(t, __shfl_xor(t, 8));
      float mn = fmaxf(m8[e], t);
      float sc = __expf(m8[e] - mn);
      float pA = __expf(vA - mn), pB = __expf(vB - mn);
      float rs = pA + pB;
      rs += __shfl_xor(rs, 1);
      rs += __shfl_xor(rs, 2);
      rs += __shfl_xor(rs, 4);
      rs += __shfl_xor(rs, 8);
      l8[e] = l8[e] * sc + rs;
      m8[e] = mn;
      o0[e] *= sc; o1[e] *= sc; o2[e] *= sc; o3[e] *= sc;
      pw[m * 32 + l15]      = pA;
      pw[m * 32 + 16 + l15] = pB;
    }
    // P (16x32) through LDS -> A-fragment, then attn @ V
    v16bf pf = load_afrag_f32(pw);
    o0 = wmma_bf16(pf, load_bfrag(vth + (size_t)0  * KK + j0, KK), o0);
    o1 = wmma_bf16(pf, load_bfrag(vth + (size_t)16 * KK + j0, KK), o1);
    o2 = wmma_bf16(pf, load_bfrag(vth + (size_t)32 * KK + j0, KK), o2);
    o3 = wmma_bf16(pf, load_bfrag(vth + (size_t)48 * KK + j0, KK), o3);
  }

  // ---- merge per-wave partials (split-K softmax merge) ----
#pragma unroll
  for (int e = 0; e < 8; ++e) {
    int m = e + hl * 8;
    if (l15 == 0) { mst[w * 16 + m] = m8[e]; lst[w * 16 + m] = l8[e]; }
    ost[(size_t)(w * 16 + m) * 64 + 0  + l15] = o0[e];
    ost[(size_t)(w * 16 + m) * 64 + 16 + l15] = o1[e];
    ost[(size_t)(w * 16 + m) * 64 + 32 + l15] = o2[e];
    ost[(size_t)(w * 16 + m) * 64 + 48 + l15] = o3[e];
  }
  __syncthreads();
  for (int idx = tid; idx < 16 * 64; idx += 256) {
    int r = idx >> 6, d = idx & 63;
    float gm = -1e30f;
#pragma unroll
    for (int w2 = 0; w2 < 8; ++w2) gm = fmaxf(gm, mst[w2 * 16 + r]);
    float ssum = 0.f, val = 0.f;
#pragma unroll
    for (int w2 = 0; w2 < 8; ++w2) {
      float sc = __expf(mst[w2 * 16 + r] - gm);
      ssum += lst[w2 * 16 + r] * sc;
      val  += ost[(size_t)(w2 * 16 + r) * 64 + d] * sc;
    }
    attnA[((size_t)b * SS + i0 + r) * DMM + h * DD + d] = (bf16_t)(val / ssum);
  }
}

// ---------------------------------------------------------------- launch

extern "C" void kernel_launch(void* const* d_in, const int* in_sizes, int n_in,
                              void* d_out, int out_size, void* d_ws, size_t ws_size,
                              hipStream_t stream) {
  const float* x    = (const float*)d_in[0];   // [S,B,DM]
  const float* mem  = (const float*)d_in[1];   // [M,B,DM]
  const float* rel  = (const float*)d_in[2];   // [K,DM]
  const int*   amsk = (const int*)d_in[3];     // [B,S,S]
  const float* Wq   = (const float*)d_in[5];
  const float* bq   = (const float*)d_in[6];
  const float* Wke  = (const float*)d_in[7];
  const float* bke  = (const float*)d_in[8];
  const float* Wv   = (const float*)d_in[9];
  const float* bv   = (const float*)d_in[10];
  const float* Wkr  = (const float*)d_in[11];
  const float* bkr  = (const float*)d_in[12];
  const float* uvec = (const float*)d_in[13];  // [H,D]
  const float* vvec = (const float*)d_in[14];
  const float* Wo   = (const float*)d_in[15];
  const float* bo   = (const float*)d_in[16];
  float* out = (float*)d_out;                  // [B,S,DM]

  char* ws = (char*)d_ws;
  size_t off = 0;
  auto alloc = [&](size_t bytes) {
    size_t r = off;
    off = (off + bytes + 255) & ~(size_t)255;
    return r;
  };
  const size_t QROWS = (size_t)SS * BB;   // 4096
  const size_t KVROWS = (size_t)KK * BB;  // 5120
  bf16_t* xin  = (bf16_t*)(ws + alloc(QROWS * DMM * 2));
  bf16_t* kv   = (bf16_t*)(ws + alloc(KVROWS * DMM * 2));
  bf16_t* relb = (bf16_t*)(ws + alloc((size_t)KK * DMM * 2));
  bf16_t* wqb  = (bf16_t*)(ws + alloc((size_t)DMM * DMM * 2));
  bf16_t* wkeb = (bf16_t*)(ws + alloc((size_t)DMM * DMM * 2));
  bf16_t* wvb  = (bf16_t*)(ws + alloc((size_t)DMM * DMM * 2));
  bf16_t* wkrb = (bf16_t*)(ws + alloc((size_t)DMM * DMM * 2));
  bf16_t* wob  = (bf16_t*)(ws + alloc((size_t)DMM * DMM * 2));
  bf16_t* q_u  = (bf16_t*)(ws + alloc(QROWS * DMM * 2));
  bf16_t* q_v  = (bf16_t*)(ws + alloc(QROWS * DMM * 2));
  bf16_t* kxb  = (bf16_t*)(ws + alloc((size_t)BB * HH * KK * DD * 2));
  bf16_t* vxT  = (bf16_t*)(ws + alloc((size_t)BB * HH * DD * KK * 2));
  bf16_t* krb  = (bf16_t*)(ws + alloc((size_t)HH * KK * DD * 2));
  bf16_t* attnA = (bf16_t*)(ws + alloc(QROWS * DMM * 2));
  (void)ws_size; (void)in_sizes; (void)n_in; (void)out_size; (void)d_in;

  auto cvt = [&](const float* s, bf16_t* d, size_t n) {
    cvt_f32_bf16<<<dim3((unsigned)((n + 255) / 256)), 256, 0, stream>>>(s, d, (int)n);
  };
  // conversions
  cvt(x, xin, QROWS * DMM);
  cvt(mem, kv, (size_t)MMEM * BB * DMM);                 // rows j*B+b, j<M
  cvt(x, kv + (size_t)MMEM * BB * DMM, QROWS * DMM);     // rows j>=M
  cvt(rel, relb, (size_t)KK * DMM);
  cvt(Wq, wqb, (size_t)DMM * DMM);
  cvt(Wke, wkeb, (size_t)DMM * DMM);
  cvt(Wv, wvb, (size_t)DMM * DMM);
  cvt(Wkr, wkrb, (size_t)DMM * DMM);
  cvt(Wo, wob, (size_t)DMM * DMM);

  // projections (one wave per 16x64 tile)
  qproj_kernel<<<dim3(QROWS / 16, DMM / 64), 32, 0, stream>>>(
      xin, wqb, bq, uvec, vvec, q_u, q_v);
  keproj_kernel<<<dim3(KVROWS / 16, DMM / 64), 32, 0, stream>>>(kv, wkeb, bke, kxb);
  vproj_kernel<<<dim3(KVROWS / 16, DMM / 64), 32, 0, stream>>>(kv, wvb, bv, vxT);
  krproj_kernel<<<dim3(KK / 16, DMM / 64), 32, 0, stream>>>(relb, wkrb, bkr, krb);

  // fused rel-shift flash attention
  size_t smem_bytes = (size_t)(16 * KK + 8 * 16 * 32 + 8 * 16 * 64 + 256) * 4;
  flash_kernel<<<dim3(SS / 16, HH, BB), 256, smem_bytes, stream>>>(
      q_u, q_v, kxb, vxT, krb, amsk, attnA);

  // output projection
  oproj_kernel<<<dim3(QROWS / 16, DMM / 64), 32, 0, stream>>>(attnA, wob, bo, out);
}
